// MultiScaleRetention_51384988729925
// MI455X (gfx1250) — compile-verified, matrix operand order checked by reading, not run on hardware
//
#include <hip/hip_runtime.h>

// MultiScaleRetention forward, fused for gfx1250 (wave32, WMMA bf16,
// double-buffered async global->LDS pipeline).
// B=1, T=4096, E=1024, VD=2048, H=8, KD=128, HD=256.

#define TT 4096
#define EE 1024
#define VDD 2048
#define HH 8
#define KDD 128
#define HDD 256

typedef __attribute__((ext_vector_type(16))) __bf16 v16bf;
typedef __attribute__((ext_vector_type(8)))  float  v8f;

union BF16Frag {
  v16bf v;
  unsigned short u[16];
  unsigned int   w[8];
};

__device__ __forceinline__ unsigned short f2bf(float f) {
  unsigned int u = __float_as_uint(f);
  u += 0x7FFFu + ((u >> 16) & 1u);          // round-to-nearest-even
  return (unsigned short)(u >> 16);
}

__device__ __forceinline__ v8f vzero8() {
  v8f z = {0.f, 0.f, 0.f, 0.f, 0.f, 0.f, 0.f, 0.f};
  return z;
}

// ---- CDNA5 async global->LDS copy (16B per lane), tracked by ASYNCcnt ----
__device__ __forceinline__ void async_b128(void* lds, const void* g) {
  unsigned int l32;
  {
    __attribute__((address_space(3))) unsigned short* lp =
        (__attribute__((address_space(3))) unsigned short*)lds;
    l32 = (unsigned int)(unsigned long long)lp;
  }
  unsigned long long g64 = (unsigned long long)(size_t)g;
  asm volatile("global_load_async_to_lds_b128 %0, %1, off"
               :: "v"(l32), "v"(g64) : "memory");
}

__device__ __forceinline__ void wait_async0() {
  asm volatile("s_wait_asynccnt 0" ::: "memory");
}

// A-matrix 16x32 bf16 fragment (ISA 7.12.2): lane m = lane&15;
// lanes 0-15 hold K = k0+{0..7, 16..23}, lanes 16-31 hold K = k0+{8..15, 24..31},
// packed bf16 pairs -> b128-mergeable loads. Also serves as the B-layout load
// when the source is stored K-contiguous per N row (i.e. B^T row-major).
__device__ __forceinline__ v16bf load_row_frag(const unsigned short* __restrict__ base,
                                               int ld, int k0) {
  const int lane = threadIdx.x & 31;
  const int m  = lane & 15;
  const int hi = lane >> 4;
  BF16Frag f;
  const unsigned short* p = base + (size_t)m * ld + k0 + hi * 8;
#pragma unroll
  for (int i = 0; i < 4; ++i) f.w[i] = *(const unsigned int*)(p + 2 * i);
  p += 16;
#pragma unroll
  for (int i = 0; i < 4; ++i) f.w[4 + i] = *(const unsigned int*)(p + 2 * i);
  return f.v;
}

// ---------------------------------------------------------------- conversions
__global__ void cvt_bf16_kernel(const float* __restrict__ src,
                                unsigned short* __restrict__ dst, long long n) {
  long long i = (long long)blockIdx.x * blockDim.x + threadIdx.x;
  const long long stride = (long long)gridDim.x * blockDim.x;
  for (; i < n; i += stride) dst[i] = f2bf(src[i]);
}

// transpose + convert: src f32 [R][C] -> dst bf16 [C][R]  (R, C multiples of 32)
__global__ void __launch_bounds__(256)
cvtT_bf16_kernel(const float* __restrict__ src, unsigned short* __restrict__ dst,
                 int R, int C) {
  __shared__ float tile[32][33];
  const int bx = blockIdx.x * 32;   // C offset
  const int by = blockIdx.y * 32;   // R offset
  const int tx = threadIdx.x & 31;
  const int ty = threadIdx.x >> 5;  // 0..7
#pragma unroll
  for (int i = 0; i < 4; ++i)
    tile[ty + i * 8][tx] = src[(size_t)(by + ty + i * 8) * C + bx + tx];
  __syncthreads();
#pragma unroll
  for (int i = 0; i < 4; ++i)
    dst[(size_t)(bx + ty + i * 8) * R + by + tx] = f2bf(tile[tx][ty + i * 8]);
}

// ---------------------------------------------------------------- GEMM (bf16 -> f32)
// C[M,N] = A[M,K] @ B[K,N], with B supplied transposed (BT: [N][K] row-major).
// 256 threads = 8 waves as 2(M) x 4(N); wave tile 32x64, block tile 64(M) x 256(N).
// 2-stage double-buffered async global->LDS pipeline: copy of k-step i+1 overlaps
// WMMA of k-step i; one barrier per step.
__global__ void __launch_bounds__(256)
gemm_bf16_kernel(const unsigned short* __restrict__ A,
                 const unsigned short* __restrict__ BT,
                 float* __restrict__ C, int M, int N, int K) {
  __shared__ __align__(16) unsigned short As[2][64][32];    //  8 KB
  __shared__ __align__(16) unsigned short Bs[2][256][32];   // 32 KB
  const int tid = threadIdx.x;
  const int wave = tid >> 5;
  const int wm = wave >> 2, wn = wave & 3;
  const int m0 = blockIdx.y * 64;
  const int n0 = blockIdx.x * 256;
  const int lane = tid & 31;
  const int nl = lane & 15, hi = lane >> 4;

  v8f acc[2][4];
#pragma unroll
  for (int mi = 0; mi < 2; ++mi)
#pragma unroll
    for (int t = 0; t < 4; ++t) acc[mi][t] = vzero8();

  auto stage = [&](int buf, int k0) {
    {  // A tile: 64 rows x 64B = 256 x b128, one per thread
      const int row = tid >> 2, ch = tid & 3;
      async_b128(&As[buf][row][ch * 8], A + (size_t)(m0 + row) * K + k0 + ch * 8);
    }
#pragma unroll
    for (int it = 0; it < 4; ++it) {  // BT panel: 256 rows x 64B = 1024 x b128
      const int id = tid + 256 * it;
      const int row = id >> 2, ch = id & 3;
      async_b128(&Bs[buf][row][ch * 8], BT + (size_t)(n0 + row) * K + k0 + ch * 8);
    }
  };

  stage(0, 0);
  int buf = 0;
  for (int k0 = 0; k0 < K; k0 += 32) {
    wait_async0();
    __syncthreads();                      // buf ready; prior reads of buf^1 drained
    if (k0 + 32 < K) stage(buf ^ 1, k0 + 32);

    v16bf a0 = load_row_frag(&As[buf][wm * 32][0], 32, 0);
    v16bf a1 = load_row_frag(&As[buf][wm * 32 + 16][0], 32, 0);
#pragma unroll
    for (int t = 0; t < 4; ++t) {
      v16bf b = load_row_frag(&Bs[buf][wn * 64 + t * 16][0], 32, 0);
      acc[0][t] = __builtin_amdgcn_wmma_f32_16x16x32_bf16(false, a0, false, b,
                                                          (short)0, acc[0][t], false, false);
      acc[1][t] = __builtin_amdgcn_wmma_f32_16x16x32_bf16(false, a1, false, b,
                                                          (short)0, acc[1][t], false, false);
    }
    buf ^= 1;
  }
#pragma unroll
  for (int mi = 0; mi < 2; ++mi)
#pragma unroll
    for (int t = 0; t < 4; ++t)
#pragma unroll
      for (int j = 0; j < 8; ++j)
        C[(size_t)(m0 + wm * 32 + mi * 16 + j + 8 * hi) * N +
          n0 + wn * 64 + t * 16 + nl] = acc[mi][t][j];
}

// ---------------------------------------------------------------- theta shift
__global__ void theta_shift_kernel(const float* __restrict__ q,
                                   const float* __restrict__ k,
                                   const float* __restrict__ sn,
                                   const float* __restrict__ cs,
                                   unsigned short* __restrict__ qrb,
                                   unsigned short* __restrict__ krb) {
  const long long idx = (long long)blockIdx.x * blockDim.x + threadIdx.x;
  const long long total = (long long)TT * EE / 2;
  if (idx >= total) return;
  const long long t = idx / (EE / 2);
  const int c = (int)(idx % (EE / 2)) * 2;
  const int d = c & (KDD - 1);
  const float s0 = sn[t * KDD + d], s1 = sn[t * KDD + d + 1];
  const float c0 = cs[t * KDD + d], c1 = cs[t * KDD + d + 1];
  const size_t off = (size_t)t * EE + c;
  const float q0 = q[off], q1 = q[off + 1];
  qrb[off]     = f2bf(q0 * c0 - q1 * s0);
  qrb[off + 1] = f2bf(q1 * c1 + q0 * s1);
  const float scale = 0.08838834764831845f;  // KD^-0.5
  const float k0v = k[off] * scale, k1v = k[off + 1] * scale;
  krb[off]     = f2bf(k0v * c0 - k1v * s0);
  krb[off + 1] = f2bf(k1v * c1 + k0v * s1);
}

// ---------------------------------------------------------------- fused retention
// grid (T/16, H), 128 threads (4 waves).
// Pass 1: denom = clip(sum_s |qk*mask|, 1, 5e4) (waves split the causal s-range).
// Pass 2: out = (qk*mask/denom) @ v with a double-buffered async-staged k-slab;
//         the first slab is prefetched before pass 1 so the copy overlaps it.
__global__ void __launch_bounds__(128)
retention_kernel(const unsigned short* __restrict__ qr,
                 const unsigned short* __restrict__ kr,
                 const unsigned short* __restrict__ vbT,
                 const float* __restrict__ mask,
                 float* __restrict__ att) {
  const int rt = blockIdx.x, h = blockIdx.y;
  const int t0 = rt * 16;
  const int wave = threadIdx.x >> 5;
  const int lane = threadIdx.x & 31;
  const int nl = lane & 15, hi = lane >> 4;

  __shared__ float part[4][16];
  __shared__ float invden[16];
  __shared__ float qkt[4][16][34];                          // per-wave 16x32 slab
  __shared__ __align__(16) unsigned short Ks[2][32][KDD];   // double-buffered k-slab

  const unsigned short* qtile = qr + (size_t)t0 * EE + h * KDD;
  const float* mbase = mask + ((size_t)h * TT + t0) * TT;
  const int npair = rt / 2 + 1;             // 16x32 slabs; overhang masked to 0

  auto stageK = [&](int buf, int p) {
    const int s0 = p * 32;
#pragma unroll
    for (int it = 0; it < 4; ++it) {        // 32 rows x 256B = 512 x b128
      const int id = threadIdx.x + 128 * it;
      const int row = id >> 4, ch = id & 15;
      async_b128(&Ks[buf][row][ch * 8],
                 kr + (size_t)(s0 + row) * EE + h * KDD + ch * 8);
    }
  };
  stageK(0, 0);                             // prefetch overlaps all of pass 1

  // ---- pass 1 ----
  float rsum[8];
#pragma unroll
  for (int j = 0; j < 8; ++j) rsum[j] = 0.f;

  for (int st = wave; st <= rt; st += 4) {
    const int s0 = st * 16;
    const unsigned short* ktile = kr + (size_t)s0 * EE + h * KDD;
    v8f acc = vzero8();
#pragma unroll
    for (int k0 = 0; k0 < KDD; k0 += 32) {
      v16bf a = load_row_frag(qtile, EE, k0);
      v16bf b = load_row_frag(ktile, EE, k0);   // kr^T: K-contiguous rows
      acc = __builtin_amdgcn_wmma_f32_16x16x32_bf16(false, a, false, b,
                                                    (short)0, acc, false, false);
    }
#pragma unroll
    for (int j = 0; j < 8; ++j) {
      const int r = j + 8 * hi;
      rsum[j] += fabsf(acc[j] * mbase[(size_t)r * TT + s0 + nl]);
    }
  }
#pragma unroll
  for (int j = 0; j < 8; ++j) {
    float s = rsum[j];
    s += __shfl_xor(s, 1, 32);
    s += __shfl_xor(s, 2, 32);
    s += __shfl_xor(s, 4, 32);
    s += __shfl_xor(s, 8, 32);
    if (nl == 0) part[wave][j + 8 * hi] = s;
  }
  __syncthreads();
  if (threadIdx.x < 16) {
    float tot = part[0][threadIdx.x] + part[1][threadIdx.x] +
                part[2][threadIdx.x] + part[3][threadIdx.x];
    tot = fminf(fmaxf(tot, 1.0f), 50000.0f);
    invden[threadIdx.x] = 1.0f / tot;
  }
  __syncthreads();

  // ---- pass 2 ----
  v8f oacc[4];
#pragma unroll
  for (int t = 0; t < 4; ++t) oacc[t] = vzero8();

  int kbuf = 0;
  for (int p = 0; p < npair; ++p) {
    wait_async0();
    __syncthreads();                        // Ks[kbuf] ready; reads of kbuf^1 drained
    if (p + 1 < npair) stageK(kbuf ^ 1, p + 1);

    const int s0 = p * 32;
#pragma unroll
    for (int half = 0; half < 2; ++half) {
      const int sb = s0 + half * 16;
      v8f acc = vzero8();
#pragma unroll
      for (int k0 = 0; k0 < KDD; k0 += 32) {
        v16bf a = load_row_frag(qtile, EE, k0);
        v16bf b = load_row_frag(&Ks[kbuf][half * 16][0], KDD, k0);
        acc = __builtin_amdgcn_wmma_f32_16x16x32_bf16(false, a, false, b,
                                                      (short)0, acc, false, false);
      }
#pragma unroll
      for (int j = 0; j < 8; ++j) {
        const int r = j + 8 * hi;
        const float mv = mbase[(size_t)r * TT + sb + nl];
        qkt[wave][r][half * 16 + nl] = acc[j] * mv * invden[r];
      }
    }
    // restage C-layout slab into A-layout bf16 fragment (wave-private LDS)
    BF16Frag af;
    const float* arow = &qkt[wave][nl][hi * 8];
#pragma unroll
    for (int i = 0; i < 4; ++i) {
      af.u[2 * i]     = f2bf(arow[2 * i]);
      af.u[2 * i + 1] = f2bf(arow[2 * i + 1]);
    }
#pragma unroll
    for (int i = 0; i < 4; ++i) {
      af.u[8 + 2 * i] = f2bf(arow[16 + 2 * i]);
      af.u[9 + 2 * i] = f2bf(arow[16 + 2 * i + 1]);
    }
    // v fragments from vbT: row (channel) contiguous in s -> K-contiguous
#pragma unroll
    for (int t = 0; t < 4; ++t) {
      const unsigned short* vbase =
          vbT + (size_t)(h * HDD + wave * 64 + t * 16) * TT + s0;
      v16bf b = load_row_frag(vbase, TT, 0);
      oacc[t] = __builtin_amdgcn_wmma_f32_16x16x32_bf16(false, af.v, false, b,
                                                        (short)0, oacc[t], false, false);
    }
    kbuf ^= 1;
  }
#pragma unroll
  for (int t = 0; t < 4; ++t)
#pragma unroll
    for (int j = 0; j < 8; ++j)
      att[(size_t)(t0 + j + 8 * hi) * VDD + h * HDD + wave * 64 + t * 16 + nl] = oacc[t][j];
}

// ---------------------------------------------------------------- RMS-norm + SiLU
__global__ void __launch_bounds__(256)
gnorm_silu_kernel(const float* __restrict__ att, const float* __restrict__ g,
                  unsigned short* __restrict__ yb) {
  const int th = blockIdx.x;
  const int t = th / HH, h = th % HH;
  const size_t off = (size_t)t * VDD + h * HDD + threadIdx.x;
  const float v = att[off];
  float s = v * v;
#pragma unroll
  for (int m = 16; m >= 1; m >>= 1) s += __shfl_xor(s, m, 32);
  __shared__ float red[8];
  __shared__ float totsh;
  if ((threadIdx.x & 31) == 0) red[threadIdx.x >> 5] = s;
  __syncthreads();
  if (threadIdx.x == 0) {
    float tt = 0.f;
#pragma unroll
    for (int i = 0; i < 8; ++i) tt += red[i];
    totsh = tt;
  }
  __syncthreads();
  const float r = rsqrtf(totsh / (float)HDD + 1e-6f);
  const float gv = g[off];
  const float sil = gv / (1.f + expf(-gv));
  yb[off] = f2bf(sil * v * r);
}

// ---------------------------------------------------------------- launch
extern "C" void kernel_launch(void* const* d_in, const int* in_sizes, int n_in,
                              void* d_out, int out_size, void* d_ws, size_t ws_size,
                              hipStream_t stream) {
  (void)in_sizes; (void)n_in; (void)out_size; (void)ws_size;
  const float* x    = (const float*)d_in[0];
  const float* sn   = (const float*)d_in[1];
  const float* cs   = (const float*)d_in[2];
  const float* mask = (const float*)d_in[3];
  const float* Wq   = (const float*)d_in[4];
  const float* Wk   = (const float*)d_in[5];
  const float* Wv   = (const float*)d_in[6];
  const float* Wg   = (const float*)d_in[7];
  const float* Wo   = (const float*)d_in[8];
  float* out = (float*)d_out;

  char* ws = (char*)d_ws;
  auto alloc = [&](size_t bytes) -> char* {
    char* p = ws;
    ws += (bytes + 255) & ~(size_t)255;
    return p;
  };
  unsigned short* xb   = (unsigned short*)alloc(sizeof(unsigned short) * TT * EE);
  unsigned short* wqbT = (unsigned short*)alloc(sizeof(unsigned short) * EE * EE);
  unsigned short* wkbT = (unsigned short*)alloc(sizeof(unsigned short) * EE * EE);
  unsigned short* wvbT = (unsigned short*)alloc(sizeof(unsigned short) * EE * VDD);
  unsigned short* wgbT = (unsigned short*)alloc(sizeof(unsigned short) * EE * VDD);
  unsigned short* wobT = (unsigned short*)alloc(sizeof(unsigned short) * VDD * EE);
  float* qf   = (float*)alloc(sizeof(float) * TT * EE);
  float* kf   = (float*)alloc(sizeof(float) * TT * EE);
  float* vf   = (float*)alloc(sizeof(float) * TT * VDD);
  float* gf   = (float*)alloc(sizeof(float) * TT * VDD);
  unsigned short* qrb = (unsigned short*)alloc(sizeof(unsigned short) * TT * EE);
  unsigned short* krb = (unsigned short*)alloc(sizeof(unsigned short) * TT * EE);
  unsigned short* vbT = (unsigned short*)alloc(sizeof(unsigned short) * TT * VDD);
  float* attf = (float*)alloc(sizeof(float) * TT * VDD);
  unsigned short* yb  = (unsigned short*)alloc(sizeof(unsigned short) * TT * VDD);

  // x -> bf16 (row-major A operand)
  cvt_bf16_kernel<<<4096, 256, 0, stream>>>(x, xb, (long long)TT * EE);
  // weights -> bf16 transposed ([N][K]) for K-contiguous B fragments
  cvtT_bf16_kernel<<<dim3(EE / 32, EE / 32), 256, 0, stream>>>(Wq, wqbT, EE, EE);
  cvtT_bf16_kernel<<<dim3(EE / 32, EE / 32), 256, 0, stream>>>(Wk, wkbT, EE, EE);
  cvtT_bf16_kernel<<<dim3(VDD / 32, EE / 32), 256, 0, stream>>>(Wv, wvbT, EE, VDD);
  cvtT_bf16_kernel<<<dim3(VDD / 32, EE / 32), 256, 0, stream>>>(Wg, wgbT, EE, VDD);
  cvtT_bf16_kernel<<<dim3(EE / 32, VDD / 32), 256, 0, stream>>>(Wo, wobT, VDD, EE);

  // projections: q,k (N=1024), v,g (N=2048); K=1024; block tile 64x256
  gemm_bf16_kernel<<<dim3(EE / 256, TT / 64), 256, 0, stream>>>(xb, wqbT, qf, TT, EE, EE);
  gemm_bf16_kernel<<<dim3(EE / 256, TT / 64), 256, 0, stream>>>(xb, wkbT, kf, TT, EE, EE);
  gemm_bf16_kernel<<<dim3(VDD / 256, TT / 64), 256, 0, stream>>>(xb, wvbT, vf, TT, VDD, EE);
  gemm_bf16_kernel<<<dim3(VDD / 256, TT / 64), 256, 0, stream>>>(xb, wgbT, gf, TT, VDD, EE);

  {
    long long pairs = (long long)TT * EE / 2;
    theta_shift_kernel<<<(int)((pairs + 255) / 256), 256, 0, stream>>>(qf, kf, sn, cs, qrb, krb);
  }
  // v -> bf16 transposed [VD][T]
  cvtT_bf16_kernel<<<dim3(VDD / 32, TT / 32), 256, 0, stream>>>(vf, vbT, TT, VDD);

  retention_kernel<<<dim3(TT / 16, HH), 128, 0, stream>>>(qrb, krb, vbT, mask, attf);

  gnorm_silu_kernel<<<TT * HH, 256, 0, stream>>>(attf, gf, yb);

  // final: out = y (T x VD) @ Wo (VD x E), K=2048
  gemm_bf16_kernel<<<dim3(EE / 256, TT / 64), 256, 0, stream>>>(yb, wobT, out, TT, EE, VDD);
}
